// DecoupledRoPEAttention_35811437314649
// MI455X (gfx1250) — matrix operand order repaired and checked by reading, not compiled, not run on hardware
//
#include <hip/hip_runtime.h>
#include <hip/hip_bf16.h>

// ---------------------------------------------------------------------------
// Problem constants (match reference)
// ---------------------------------------------------------------------------
static constexpr int BB   = 2;
static constexpr int MM   = 2048;
static constexpr int NN   = 2048;
static constexpr int DD   = 2048;
static constexpr int NH   = 16;
static constexpr int DH   = 128;
static constexpr int DR   = 64;
static constexpr float SCALE = 0.07216878364870323f;  // 1/sqrt(192)
static constexpr float LN_THETA = 13.815510557964274f; // ln(1e6)

typedef __attribute__((ext_vector_type(16))) __bf16 v16bf;
typedef __attribute__((ext_vector_type(8)))  __bf16 v8bf;
typedef __attribute__((ext_vector_type(8)))  float  v8f;
typedef int v4i_t __attribute__((vector_size(16)));   // GCC-style int4 (matches builtin proto)

// ---------------------------------------------------------------------------
// WMMA helper (bf16 A,B -> f32 accumulate), guarded so host pass parses clean
// ---------------------------------------------------------------------------
__device__ __forceinline__ v8f wmma_bf16(v16bf a, v16bf b, v8f c) {
#if defined(__gfx1250__)
  return __builtin_amdgcn_wmma_f32_16x16x32_bf16(
      /*neg_a=*/false, a, /*neg_b=*/false, b,
      /*c_mod=*/(short)0, c, /*reuse_a=*/false, /*reuse_b=*/false);
#else
  (void)a; (void)b;
  return c;
#endif
}

// ---------------------------------------------------------------------------
// Async global->LDS copies (CDNA5 ASYNCcnt path), with sync fallback
// ---------------------------------------------------------------------------
#if defined(__gfx1250__) && __has_builtin(__builtin_amdgcn_global_load_async_to_lds_b128)
#define USE_ASYNC_LDS 1
__device__ __forceinline__ void async_copy_b128(const __bf16* g, __bf16* l) {
  __builtin_amdgcn_global_load_async_to_lds_b128(
      (__attribute__((address_space(1))) v4i_t*)g,
      (__attribute__((address_space(3))) v4i_t*)l,
      /*offset=*/0, /*cpol=*/0);
}
__device__ __forceinline__ void wait_async0() {
#if __has_builtin(__builtin_amdgcn_s_wait_asynccnt)
  __builtin_amdgcn_s_wait_asynccnt(0);
#else
  asm volatile("s_wait_asynccnt 0" ::: "memory");
#endif
}
#endif

// A-fragment (16x32, 16-bit): lane l (l15=l&15) holds row M=l15;
// kbase = (l<16)?0:16 ; elements 0..15 = K kbase..kbase+15 (contiguous 32B).
__device__ __forceinline__ v16bf load_afrag(const __bf16* p) {
  const v8bf lo = *reinterpret_cast<const v8bf*>(p);
  const v8bf hi = *reinterpret_cast<const v8bf*>(p + 8);
  v16bf r;
#pragma unroll
  for (int i = 0; i < 8; ++i) { r[i] = lo[i]; r[i + 8] = hi[i]; }
  return r;
}

// B-fragment (32x16, 16-bit): lane l holds column N=l&15;
// kbase = (l<16)?0:16 ; elements 0..15 = K kbase..kbase+15 (contiguous 32B).
__device__ __forceinline__ v16bf load_bfrag(const __bf16* p) {
  const v8bf lo = *reinterpret_cast<const v8bf*>(p);
  const v8bf hi = *reinterpret_cast<const v8bf*>(p + 8);
  v16bf r;
#pragma unroll
  for (int i = 0; i < 8; ++i) { r[i] = lo[i]; r[i + 8] = hi[i]; }
  return r;
}

// ---------------------------------------------------------------------------
// fp32 -> bf16 conversion
// ---------------------------------------------------------------------------
__global__ void f32_to_bf16_kernel(const float* __restrict__ src,
                                   __bf16* __restrict__ dst, size_t n) {
  size_t i = (size_t)blockIdx.x * blockDim.x + threadIdx.x;
  const size_t stride = (size_t)gridDim.x * blockDim.x;
  for (; i < n; i += stride) dst[i] = (__bf16)src[i];
}

// ---------------------------------------------------------------------------
// bf16 GEMM:  Y[m,o] = sum_k X[m,k] * W[o,k]   (i.e. Y = X @ W^T)
// Block tile 128x128, BK=32, double-buffered LDS staged with async-to-LDS.
// 8 waves; wave (wm,wn) computes a 64x32 sub-tile = 4x2 WMMA accumulators.
// ---------------------------------------------------------------------------
template <bool OUT_F32>
__global__ __launch_bounds__(256) void gemm_bf16_kernel(
    const __bf16* __restrict__ X, const __bf16* __restrict__ W,
    void* __restrict__ Yv, int Mrows, int O, int K) {
  __shared__ __bf16 Xs[2][128 * 40];  // stride 40 to dodge bank conflicts
  __shared__ __bf16 Ws[2][128 * 40];

  const int m0 = blockIdx.x * 128;
  const int o0 = blockIdx.y * 128;
  const int tid  = threadIdx.x;
  const int wave = tid >> 5;
  const int lane = tid & 31;
  const int wm = wave >> 2;       // 0..1
  const int wn = wave & 3;        // 0..3
  const int l15  = lane & 15;
  const int akb  = (lane < 16) ? 0 : 16;
  const int bkb  = (lane < 16) ? 0 : 16;
  const int rhalf = (lane < 16) ? 0 : 8;

  const int ldrow = tid >> 1;           // 0..127
  const int ldcol = (tid & 1) * 16;     // 0 / 16
  const int lofs  = ldrow * 40 + ldcol;
  const __bf16* gx = &X[(size_t)(m0 + ldrow) * K + ldcol];
  const __bf16* gw = &W[(size_t)(o0 + ldrow) * K + ldcol];

  v8f acc[4][2] = {};

  const int T = K >> 5;  // number of 32-wide K tiles

  auto stage = [&](int buf, int kt) {
#ifdef USE_ASYNC_LDS
    async_copy_b128(gx + (size_t)kt * 32, &Xs[buf][lofs]);
    async_copy_b128(gw + (size_t)kt * 32, &Ws[buf][lofs]);
#else
    *reinterpret_cast<uint4*>(&Xs[buf][lofs]) =
        *reinterpret_cast<const uint4*>(gx + (size_t)kt * 32);
    *reinterpret_cast<uint4*>(&Ws[buf][lofs]) =
        *reinterpret_cast<const uint4*>(gw + (size_t)kt * 32);
#endif
  };

  stage(0, 0);
  for (int kt = 0; kt < T; ++kt) {
    const int buf = kt & 1;
#ifdef USE_ASYNC_LDS
    wait_async0();               // this wave's tile-kt copies landed in LDS
#endif
    __syncthreads();             // every wave's tile kt visible
    if (kt + 1 < T) stage(buf ^ 1, kt + 1);  // overlap next copies w/ compute

    v16bf a[4], b[2];
#pragma unroll
    for (int i = 0; i < 4; ++i)
      a[i] = load_afrag(&Xs[buf][(wm * 64 + i * 16 + l15) * 40 + akb]);
#pragma unroll
    for (int j = 0; j < 2; ++j)
      b[j] = load_bfrag(&Ws[buf][(wn * 32 + j * 16 + l15) * 40 + bkb]);
#pragma unroll
    for (int i = 0; i < 4; ++i)
#pragma unroll
      for (int j = 0; j < 2; ++j)
        acc[i][j] = wmma_bf16(a[i], b[j], acc[i][j]);
    __syncthreads();             // safe to overwrite buf next round
  }

  // C layout: lanes 0-15: VGPR v -> (M=v, N=lane); lanes 16-31: (M=v+8, N=lane-16)
#pragma unroll
  for (int i = 0; i < 4; ++i)
#pragma unroll
    for (int j = 0; j < 2; ++j)
#pragma unroll
      for (int v = 0; v < 8; ++v) {
        const int row = m0 + wm * 64 + i * 16 + rhalf + v;
        const int col = o0 + wn * 32 + j * 16 + l15;
        if (OUT_F32)
          reinterpret_cast<float*>(Yv)[(size_t)row * O + col] = acc[i][j][v];
        else
          reinterpret_cast<__bf16*>(Yv)[(size_t)row * O + col] = (__bf16)acc[i][j][v];
      }
}

// ---------------------------------------------------------------------------
// RoPE (in place, bf16 storage, fp32 math). x: (B*seq) x cols, heads of dr.
// ---------------------------------------------------------------------------
__global__ void rope_kernel(__bf16* __restrict__ x, int cols, int dr, int seq) {
  const size_t idx = (size_t)blockIdx.x * blockDim.x + threadIdx.x;
  const int ppr = cols / 2;
  const size_t npairs = (size_t)BB * seq * ppr;
  if (idx >= npairs) return;
  const size_t row = idx / ppr;
  const int pr = (int)(idx % ppr);
  const int hp = pr / (dr / 2);
  const int i  = pr % (dr / 2);
  const int pos = (int)(row % seq);
  const float freq = __expf(-LN_THETA * (float)(2 * i) / (float)dr);
  const float ang = (float)pos * freq;
  float sn, cs;
  __sincosf(ang, &sn, &cs);
  const size_t base = row * (size_t)cols + (size_t)hp * dr + 2 * i;
  const float x0 = (float)x[base];
  const float x1 = (float)x[base + 1];
  x[base]     = (__bf16)(x0 * cs - x1 * sn);
  x[base + 1] = (__bf16)(x1 * cs + x0 * sn);
}

// ---------------------------------------------------------------------------
// Per-batch 2048x2048 bf16 transpose: Vt[b, c, n] = V[b, n, c]
// ---------------------------------------------------------------------------
__global__ __launch_bounds__(256) void transpose2048_kernel(
    const __bf16* __restrict__ Vin, __bf16* __restrict__ Vt) {
  __shared__ __bf16 tile[32][33];
  const int n0 = blockIdx.x * 32;
  const int c0 = blockIdx.y * 32;
  const int b  = blockIdx.z;
  const int tr = threadIdx.x >> 5;   // 0..7
  const int tc = threadIdx.x & 31;   // 0..31
#pragma unroll
  for (int i = 0; i < 32; i += 8)
    tile[tr + i][tc] = Vin[((size_t)(b * NN + n0 + tr + i)) * DD + c0 + tc];
  __syncthreads();
#pragma unroll
  for (int i = 0; i < 32; i += 8)
    Vt[((size_t)(b * DD + c0 + tr + i)) * NN + n0 + tc] = tile[tc][tr + i];
}

// ---------------------------------------------------------------------------
// Flash attention. grid = (B*NH, M/64), block = 128 (4 waves).
// Wave w handles Q rows [m0 + w*16, +16). KV streamed in chunks of 32.
// ---------------------------------------------------------------------------
__global__ __launch_bounds__(128) void attn_kernel(
    const __bf16* __restrict__ QC, const __bf16* __restrict__ KC,
    const __bf16* __restrict__ QR, const __bf16* __restrict__ KR,
    const __bf16* __restrict__ Vt, __bf16* __restrict__ CTX) {
  const int bh = blockIdx.x;
  const int b = bh >> 4;
  const int h = bh & 15;
  const int tid  = threadIdx.x;
  const int wave = tid >> 5;
  const int lane = tid & 31;
  const int l15  = lane & 15;
  const int akb  = (lane < 16) ? 0 : 16;
  const int bkb  = (lane < 16) ? 0 : 16;
  const int rhalf = (lane < 16) ? 0 : 8;
  const int m0 = blockIdx.y * 64 + wave * 16;

  __shared__ __bf16 Ps[4][16 * 40];   // per-wave P tile, 16x32 (stride 40)

  // Q fragments held in registers for the whole KV sweep
  v16bf qc[4], qr[2];
  {
    const size_t qrowc = ((size_t)(b * MM + m0 + l15)) * (size_t)(NH * DH) + (size_t)h * DH;
#pragma unroll
    for (int kc = 0; kc < 4; ++kc) qc[kc] = load_afrag(&QC[qrowc + kc * 32 + akb]);
    const size_t qrowr = ((size_t)(b * MM + m0 + l15)) * (size_t)(NH * DR) + (size_t)h * DR;
#pragma unroll
    for (int kc = 0; kc < 2; ++kc) qr[kc] = load_afrag(&QR[qrowr + kc * 32 + akb]);
  }

  v8f o[8] = {};
  float mrow[8], lsum[8];
#pragma unroll
  for (int v = 0; v < 8; ++v) { mrow[v] = -3.0e38f; lsum[v] = 0.0f; }

  for (int j = 0; j < NN; j += 32) {
    // --- scores: two 16x16 tiles covering KV columns [j, j+32) ---
    v8f s[2] = {};
#pragma unroll
    for (int nt = 0; nt < 2; ++nt) {
      const size_t krowc = ((size_t)(b * NN + j + nt * 16 + l15)) * (size_t)(NH * DH) + (size_t)h * DH;
#pragma unroll
      for (int kc = 0; kc < 4; ++kc) {
        const v16bf kb = load_bfrag(&KC[krowc + kc * 32 + bkb]);
        s[nt] = wmma_bf16(qc[kc], kb, s[nt]);
      }
      const size_t krowr = ((size_t)(b * NN + j + nt * 16 + l15)) * (size_t)(NH * DR) + (size_t)h * DR;
#pragma unroll
      for (int kc = 0; kc < 2; ++kc) {
        const v16bf kb = load_bfrag(&KR[krowr + kc * 32 + bkb]);
        s[nt] = wmma_bf16(qr[kc], kb, s[nt]);
      }
    }

    // --- online softmax (row stats via 16-lane half-wave reductions) ---
#pragma unroll
    for (int v = 0; v < 8; ++v) {
      const float s0 = s[0][v] * SCALE;
      const float s1 = s[1][v] * SCALE;
      float t = fmaxf(s0, s1);
#pragma unroll
      for (int off = 1; off < 16; off <<= 1) t = fmaxf(t, __shfl_xor(t, off, 16));
      const float mn = fmaxf(mrow[v], t);
      const float alpha = __expf(mrow[v] - mn);
      const float p0 = __expf(s0 - mn);
      const float p1 = __expf(s1 - mn);
      float rs = p0 + p1;
#pragma unroll
      for (int off = 1; off < 16; off <<= 1) rs += __shfl_xor(rs, off, 16);
      lsum[v] = lsum[v] * alpha + rs;
      mrow[v] = mn;
#pragma unroll
      for (int od = 0; od < 8; ++od) o[od][v] *= alpha;
      const int prow = v + rhalf;
      Ps[wave][prow * 40 + l15]      = (__bf16)p0;
      Ps[wave][prow * 40 + 16 + l15] = (__bf16)p1;
    }
    __syncthreads();

    // --- O += P(16x32) @ V(32x128): 8 WMMAs across D_H ---
    const v16bf pa = load_afrag(&Ps[wave][l15 * 40 + akb]);
#pragma unroll
    for (int od = 0; od < 8; ++od) {
      const size_t vtrow =
          ((size_t)(b * DD + h * DH + od * 16 + l15)) * (size_t)NN + j + bkb;
      const v16bf vb = load_bfrag(&Vt[vtrow]);
      o[od] = wmma_bf16(pa, vb, o[od]);
    }
  }

  // --- normalize and store context (bf16) ---
#pragma unroll
  for (int od = 0; od < 8; ++od)
#pragma unroll
    for (int v = 0; v < 8; ++v) {
      const int row = m0 + v + rhalf;
      const int col = h * DH + od * 16 + l15;
      CTX[((size_t)(b * MM + row)) * (size_t)(NH * DH) + col] =
          (__bf16)(o[od][v] / lsum[v]);
    }
}

// ---------------------------------------------------------------------------
// Host-side launcher
// ---------------------------------------------------------------------------
extern "C" void kernel_launch(void* const* d_in, const int* in_sizes, int n_in,
                              void* d_out, int out_size, void* d_ws, size_t ws_size,
                              hipStream_t stream) {
  (void)in_sizes; (void)n_in; (void)out_size; (void)ws_size;

  const float* query = (const float*)d_in[0];
  const float* keyv  = (const float*)d_in[1];
  const float* wQC   = (const float*)d_in[2];
  const float* wKC   = (const float*)d_in[3];
  const float* wQR   = (const float*)d_in[4];
  const float* wKR   = (const float*)d_in[5];
  const float* wV    = (const float*)d_in[6];
  const float* wO    = (const float*)d_in[7];
  float* out = (float*)d_out;

  const size_t N_ACT  = (size_t)BB * MM * DD;          // 8388608
  const size_t N_W128 = (size_t)NH * DH * DD;          // 4194304
  const size_t N_W64  = (size_t)NH * DR * DD;          // 2097152
  const size_t N_QR   = (size_t)BB * MM * NH * DR;     // 4194304

  char* ws = (char*)d_ws;
  auto alloc = [&](size_t elems) -> __bf16* {
    __bf16* p = (__bf16*)ws;
    ws += ((elems * sizeof(__bf16) + 255) / 256) * 256;
    return p;
  };
  __bf16* qB  = alloc(N_ACT);
  __bf16* kvB = alloc(N_ACT);
  __bf16* bQC = alloc(N_W128);
  __bf16* bKC = alloc(N_W128);
  __bf16* bQR = alloc(N_W64);
  __bf16* bKR = alloc(N_W64);
  __bf16* bV  = alloc(N_W128);
  __bf16* bO  = alloc(N_W128);
  __bf16* QC  = alloc(N_ACT);
  __bf16* KC  = alloc(N_ACT);
  __bf16* QR  = alloc(N_QR);
  __bf16* KR  = alloc(N_QR);
  __bf16* Vb  = alloc(N_ACT);
  // Aliased buffers (producers of qB / kvB are fully consumed before these
  // are written, in stream order):
  __bf16* Vt  = kvB;   // written by transpose after the last kvB-reading GEMM
  __bf16* CTX = qB;    // written by attention after the last qB-reading GEMM

  const int CT = 256;
  const int CG = 8192;
  f32_to_bf16_kernel<<<CG, CT, 0, stream>>>(query, qB, N_ACT);
  f32_to_bf16_kernel<<<CG, CT, 0, stream>>>(keyv, kvB, N_ACT);
  f32_to_bf16_kernel<<<CG, CT, 0, stream>>>(wQC, bQC, N_W128);
  f32_to_bf16_kernel<<<CG, CT, 0, stream>>>(wKC, bKC, N_W128);
  f32_to_bf16_kernel<<<CG, CT, 0, stream>>>(wQR, bQR, N_W64);
  f32_to_bf16_kernel<<<CG, CT, 0, stream>>>(wKR, bKR, N_W64);
  f32_to_bf16_kernel<<<CG, CT, 0, stream>>>(wV,  bV,  N_W128);
  f32_to_bf16_kernel<<<CG, CT, 0, stream>>>(wO,  bO,  N_W128);

  const int MR = BB * MM;  // 4096 rows
  // Projections (bf16 out)
  gemm_bf16_kernel<false><<<dim3(MR / 128, 2048 / 128), 256, 0, stream>>>(qB,  bQC, QC, MR, 2048, DD);
  gemm_bf16_kernel<false><<<dim3(MR / 128, 2048 / 128), 256, 0, stream>>>(kvB, bKC, KC, MR, 2048, DD);
  gemm_bf16_kernel<false><<<dim3(MR / 128, 1024 / 128), 256, 0, stream>>>(qB,  bQR, QR, MR, 1024, DD);
  gemm_bf16_kernel<false><<<dim3(MR / 128, 1024 / 128), 256, 0, stream>>>(kvB, bKR, KR, MR, 1024, DD);
  gemm_bf16_kernel<false><<<dim3(MR / 128, 2048 / 128), 256, 0, stream>>>(kvB, bV,  Vb, MR, 2048, DD);

  // RoPE on Q_R / K_R (in place)
  {
    const size_t npairs = (size_t)MR * (NH * DR / 2);   // 2,097,152
    const int blocks = (int)((npairs + CT - 1) / CT);
    rope_kernel<<<blocks, CT, 0, stream>>>(QR, NH * DR, DR, MM);
    rope_kernel<<<blocks, CT, 0, stream>>>(KR, NH * DR, DR, NN);
  }

  // V -> Vt (B, NH*DH, N)   [Vt aliases kvB, safe after the V GEMM]
  transpose2048_kernel<<<dim3(NN / 32, DD / 32, BB), 256, 0, stream>>>(Vb, Vt);

  // Flash attention -> CTX (aliases qB)
  attn_kernel<<<dim3(BB * NH, MM / 64), 128, 0, stream>>>(QC, KC, QR, KR, Vt, CTX);

  // Output projection (fp32 out)
  gemm_bf16_kernel<true><<<dim3(MR / 128, 2048 / 128), 256, 0, stream>>>(CTX, bO, out, MR, 2048, DD);
}
